// ConstrainedAttention_52527450030359
// MI455X (gfx1250) — matrix operand order repaired and checked by reading, not compiled, not run on hardware
//
#include <hip/hip_runtime.h>

// ---------------------------------------------------------------------------
// Types for CDNA5 WMMA (wave32)
// ---------------------------------------------------------------------------
typedef __attribute__((ext_vector_type(16))) __bf16 v16bf;
typedef __attribute__((ext_vector_type(8)))  __bf16 v8bf;
typedef __attribute__((ext_vector_type(8)))  float  v8f;

#define D_MODEL 1024
#define N_HEADS 16
#define D_HEAD  64
#define BATCH   2
#define SEQ     2048
#define TOKENS  (BATCH * SEQ)          // 4096
#define QKV_N   (3 * D_MODEL)          // 3072

__device__ __forceinline__ unsigned short f2bf(float f) {
    unsigned int u = __float_as_uint(f);
    u += 0x7FFFu + ((u >> 16) & 1u);   // round-to-nearest-even
    return (unsigned short)(u >> 16);
}

__device__ __forceinline__ __bf16 bf16_one() {
    unsigned short u = 0x3F80;
    __bf16 r;
    __builtin_memcpy(&r, &u, 2);
    return r;
}

// D = A*B + C  (16x16x32 bf16 -> f32)
__device__ __forceinline__ v8f wmma_bf(v16bf a, v16bf b, v8f c) {
    return __builtin_amdgcn_wmma_f32_16x16x32_bf16(
        /*neg_a=*/false, a, /*neg_b=*/false, b,
        /*c_mod=*/(short)0, c, /*reuse_a=*/false, /*reuse_b=*/false);
}

// A fragment: 16x32 bf16, rows row0..row0+15, K = k0..k0+31, leading dim ld.
// Lanes 0-15: row=lane,   K = {k0..k0+7, k0+16..k0+23}
// Lanes16-31: row=lane-16,K = {k0+8..k0+15, k0+24..k0+31}
__device__ __forceinline__ v16bf load_a(const unsigned short* base, int ld,
                                        int row0, int k0, int lane) {
    int r   = row0 + (lane & 15);
    int sel = (lane & 16) ? 8 : 0;
    const unsigned short* p = base + (size_t)r * ld + k0 + sel;
    v8bf lo = *reinterpret_cast<const v8bf*>(p);
    v8bf hi = *reinterpret_cast<const v8bf*>(p + 16);
    return __builtin_shufflevector(lo, hi, 0,1,2,3,4,5,6,7,8,9,10,11,12,13,14,15);
}

// B fragment: 32x16 bf16, columns col0..col0+15, K = k0..k0+31.
// Column c's K values contiguous with stride ld.
// Lanes 0-15: col=lane,    K = {k0..k0+7, k0+16..k0+23}
// Lanes16-31: col=lane-16, K = {k0+8..k0+15, k0+24..k0+31}
__device__ __forceinline__ v16bf load_b(const unsigned short* base, int ld,
                                        int col0, int k0, int lane) {
    int c  = col0 + (lane & 15);
    int kk = k0 + ((lane & 16) ? 8 : 0);
    const unsigned short* p = base + (size_t)c * ld + kk;
    v8bf lo = *reinterpret_cast<const v8bf*>(p);
    v8bf hi = *reinterpret_cast<const v8bf*>(p + 16);
    return __builtin_shufflevector(lo, hi, 0,1,2,3,4,5,6,7,8,9,10,11,12,13,14,15);
}

// Async Global -> LDS copy, 16 B per lane (CDNA5 GLOBAL_LOAD_ASYNC_TO_LDS).
// VDST = per-lane LDS byte offset (generic LDS pointer truncated to 32 bits,
// whose low 32 bits give the LDS address); VADDR = 64-bit global addr.
// Tracked by ASYNCcnt; pair with s_wait_asynccnt before the chunk barrier.
__device__ __forceinline__ void async_copy16B(unsigned short* lds_dst,
                                              const unsigned short* gsrc) {
    unsigned int       ldsoff = (unsigned int)(unsigned long long)(void*)lds_dst;
    unsigned long long gaddr  = (unsigned long long)(const void*)gsrc;
    asm volatile("global_load_async_to_lds_b128 %0, %1, off"
                 :: "v"(ldsoff), "v"(gaddr)
                 : "memory");
}

__device__ __forceinline__ void wait_async0() {
    asm volatile("s_wait_asynccnt 0x0" ::: "memory");
}

// ---------------------------------------------------------------------------
// Kernel 1: fp32 -> bf16 conversion of x, qkv_w, out_w
// ---------------------------------------------------------------------------
__global__ __launch_bounds__(256) void convert_kernel(
        const float* __restrict__ x, const float* __restrict__ qw,
        const float* __restrict__ ow,
        unsigned short* __restrict__ xb, unsigned short* __restrict__ qwb,
        unsigned short* __restrict__ owb) {
    const int n1 = TOKENS * D_MODEL;   // 4194304
    const int n2 = QKV_N * D_MODEL;    // 3145728
    const int n3 = D_MODEL * D_MODEL;  // 1048576
    int i = blockIdx.x * blockDim.x + threadIdx.x;
    if (i < n1) { xb[i] = f2bf(x[i]); return; }
    int j = i - n1;
    if (j < n2) { qwb[j] = f2bf(qw[j]); return; }
    int k = j - n2;
    if (k < n3) { owb[k] = f2bf(ow[k]); }
}

// ---------------------------------------------------------------------------
// Kernel 2: QKV GEMM  C[4096,3072] = Xb[4096,1024] * Wb^T + bias
// One wave -> 64x64 output tile: 16 WMMAs per 8 fragment loads (1:1 issue).
// Scatter epilogue into q / k / v^T (bf16).
// q,k layout: [b*H+h][s][d]   vt layout: [b*H+h][d][s]
// ---------------------------------------------------------------------------
__device__ __forceinline__ void store_qkv_tile(v8f c, int M0, int N0, int lane,
                                               const float* __restrict__ bias,
                                               unsigned short* __restrict__ qb,
                                               unsigned short* __restrict__ kb,
                                               unsigned short* __restrict__ vt) {
    int col   = N0 + (lane & 15);
    int hi8   = (lane & 16) ? 8 : 0;
    int which = col >> 10;            // 0=q 1=k 2=v
    int rem   = col & 1023;
    int h     = rem >> 6;
    int d     = rem & 63;
    float bv  = bias[col];
#pragma unroll
    for (int j = 0; j < 8; ++j) {
        int row = M0 + j + hi8;
        int b   = row >> 11;
        int s   = row & 2047;
        unsigned short v = f2bf(c[j] + bv);
        size_t bh = (size_t)(b * N_HEADS + h);
        if (which == 0)      qb[(bh * SEQ + s) * D_HEAD + d] = v;
        else if (which == 1) kb[(bh * SEQ + s) * D_HEAD + d] = v;
        else                 vt[(bh * D_HEAD + d) * SEQ + s] = v;
    }
}

__global__ __launch_bounds__(256) void gemm_qkv_kernel(
        const unsigned short* __restrict__ xb,
        const unsigned short* __restrict__ wb,
        const float* __restrict__ bias,
        unsigned short* __restrict__ qb, unsigned short* __restrict__ kb,
        unsigned short* __restrict__ vt) {
    const int lane  = threadIdx.x & 31;
    const int gwave = (blockIdx.x * blockDim.x + threadIdx.x) >> 5;
    const int NT = QKV_N / 64;         // 48
    const int mt = gwave / NT;
    const int nt = gwave % NT;
    const int M0 = mt * 64, N0 = nt * 64;

    v8f acc[16];
#pragma unroll
    for (int i = 0; i < 16; ++i) acc[i] = {};

#pragma unroll 2
    for (int k0 = 0; k0 < D_MODEL; k0 += 32) {
        v16bf af[4], bf[4];
#pragma unroll
        for (int i = 0; i < 4; ++i)
            af[i] = load_a(xb, D_MODEL, M0 + 16 * i, k0, lane);
#pragma unroll
        for (int i = 0; i < 4; ++i)
            bf[i] = load_b(wb, D_MODEL, N0 + 16 * i, k0, lane);
#pragma unroll
        for (int i = 0; i < 4; ++i)
#pragma unroll
            for (int j = 0; j < 4; ++j)
                acc[i * 4 + j] = wmma_bf(af[i], bf[j], acc[i * 4 + j]);
    }
#pragma unroll
    for (int i = 0; i < 4; ++i)
#pragma unroll
        for (int j = 0; j < 4; ++j)
            store_qkv_tile(acc[i * 4 + j], M0 + 16 * i, N0 + 16 * j,
                           lane, bias, qb, kb, vt);
}

// ---------------------------------------------------------------------------
// Kernel 3: flash attention, workgroup-cooperative.
// One 256-thread block = one (b,h) x 128 Q rows (8 waves x 16-row blocks).
// K/V chunks (32 cols) staged into LDS with ASYNC global->LDS copies,
// double-buffered, one s_wait_asynccnt + barrier per chunk: cuts K/V L2
// traffic 8x vs per-wave reads and skips the VGPR landing zone entirely.
// P round-trips through per-wave LDS to convert C-layout -> A-fragment
// layout; row-sums via WMMA against an all-ones B fragment.
// ---------------------------------------------------------------------------
__global__ __launch_bounds__(256) void attn_kernel(
        const unsigned short* __restrict__ q,
        const unsigned short* __restrict__ k,
        const unsigned short* __restrict__ vt,
        unsigned short* __restrict__ ctx) {
    __shared__ unsigned short sK[2][32 * D_HEAD];   // 2 x 4 KiB, rows [t][d]
    __shared__ unsigned short sV[2][D_HEAD * 32];   // 2 x 4 KiB, rows [d][t]
    __shared__ unsigned short sP[8 * 16 * 32];      // 8 KiB, per-wave P tiles

    const int tid  = threadIdx.x;
    const int lane = tid & 31;
    const int wib  = tid >> 5;
    const int bh   = blockIdx.x >> 4;               // 16 row-groups per (b,h)
    const int rg   = blockIdx.x & 15;
    const int b = bh >> 4, h = bh & 15;
    const int qrow0 = rg * 128 + wib * 16;
    unsigned short* myP = sP + wib * 512;

    const unsigned short* qbase = q  + ((size_t)bh * SEQ + qrow0) * D_HEAD;
    const unsigned short* kbase = k  + (size_t)bh * SEQ * D_HEAD;
    const unsigned short* vbase = vt + (size_t)bh * D_HEAD * SEQ;

    // Q fragments (16 rows x 64, two K-slabs) loaded once per wave.
    v16bf aq0 = load_a(qbase, D_HEAD, 0, 0,  lane);
    v16bf aq1 = load_a(qbase, D_HEAD, 0, 32, lane);

    v16bf bones;
#pragma unroll
    for (int i = 0; i < 16; ++i) bones[i] = bf16_one();

    v8f o[4]; o[0] = {}; o[1] = {}; o[2] = {}; o[3] = {};
    v8f lsum = {};
    float m[8];
#pragma unroll
    for (int j = 0; j < 8; ++j) m[j] = -1.0e30f;

    const int hi8 = (lane & 16) ? 8 : 0;
    const int cc  = lane & 15;
    const float scale = 0.125f;                     // 1/sqrt(64)

    // cooperative async stage of one 32-column K/V chunk into buffer `buf`
    // K chunk: rows t0..t0+31 contiguous (32*64 bf16 = 4 KiB flat copy)
    // V chunk: 64 rows of 32 bf16, stride SEQ; 16 B per thread each
    const int vd = tid >> 2, vp = tid & 3;
    auto stage = [&](int buf, int t0) {
        async_copy16B(&sK[buf][tid * 8], kbase + (size_t)t0 * D_HEAD + tid * 8);
        async_copy16B(&sV[buf][vd * 32 + vp * 8],
                      vbase + (size_t)vd * SEQ + t0 + vp * 8);
    };

    stage(0, 0);
    wait_async0();
    __syncthreads();

    for (int t0 = 0; t0 < SEQ; t0 += 32) {
        const int cur = (t0 >> 5) & 1;
        if (t0 + 32 < SEQ) stage(cur ^ 1, t0 + 32);

        // ---- scores: two 16x16 tiles over chunk columns 0..31 ----
        v8f s0 = {}, s1 = {};
        s0 = wmma_bf(aq0, load_b(sK[cur], D_HEAD, 0,  0,  lane), s0);
        s0 = wmma_bf(aq1, load_b(sK[cur], D_HEAD, 0,  32, lane), s0);
        s1 = wmma_bf(aq0, load_b(sK[cur], D_HEAD, 16, 0,  lane), s1);
        s1 = wmma_bf(aq1, load_b(sK[cur], D_HEAD, 16, 32, lane), s1);

        // ---- online softmax: row max, rescale, exp, stash P in LDS ----
#pragma unroll
        for (int j = 0; j < 8; ++j) {
            float a0 = s0[j] * scale;
            float a1 = s1[j] * scale;
            float v  = fmaxf(a0, a1);
            v = fmaxf(v, __shfl_xor(v, 1, 16));
            v = fmaxf(v, __shfl_xor(v, 2, 16));
            v = fmaxf(v, __shfl_xor(v, 4, 16));
            v = fmaxf(v, __shfl_xor(v, 8, 16));
            float mn   = fmaxf(m[j], v);
            float corr = __expf(m[j] - mn);
            m[j] = mn;
            o[0][j] *= corr; o[1][j] *= corr;
            o[2][j] *= corr; o[3][j] *= corr;
            lsum[j] *= corr;
            int row = j + hi8;
            myP[row * 32 + cc]      = f2bf(__expf(a0 - mn));
            myP[row * 32 + cc + 16] = f2bf(__expf(a1 - mn));
        }
        // ---- P back as A fragment (wave-private, DScnt-ordered) ----
        v16bf ap = load_a(myP, 32, 0, 0, lane);
        lsum = wmma_bf(ap, bones, lsum);            // row-sums on matrix pipe
        // ---- O += P * V (chunk-local columns of V^T in LDS) ----
        o[0] = wmma_bf(ap, load_b(sV[cur], 32, 0,  0, lane), o[0]);
        o[1] = wmma_bf(ap, load_b(sV[cur], 32, 16, 0, lane), o[1]);
        o[2] = wmma_bf(ap, load_b(sV[cur], 32, 32, 0, lane), o[2]);
        o[3] = wmma_bf(ap, load_b(sV[cur], 32, 48, 0, lane), o[3]);

        wait_async0();     // this wave's async writes to next buffer done
        __syncthreads();   // all waves' writes visible; cur free for reuse
    }

    // ---- normalize and write ctx (bf16, [b][s][h*64+d]) ----
#pragma unroll
    for (int j = 0; j < 8; ++j) {
        float invl = 1.0f / lsum[j];
        int srow = qrow0 + j + hi8;
        size_t base = ((size_t)b * SEQ + srow) * D_MODEL + h * D_HEAD + cc;
        ctx[base + 0 ] = f2bf(o[0][j] * invl);
        ctx[base + 16] = f2bf(o[1][j] * invl);
        ctx[base + 32] = f2bf(o[2][j] * invl);
        ctx[base + 48] = f2bf(o[3][j] * invl);
    }
}

// ---------------------------------------------------------------------------
// Kernel 4: output GEMM  out[4096,1024] = ctx * out_w^T + out_b   (fp32 out)
// 64x64 tile per wave, same 1:1 load:wmma inner loop.
// ---------------------------------------------------------------------------
__device__ __forceinline__ void store_out_tile(v8f c, int M0, int N0, int lane,
                                               const float* __restrict__ bias,
                                               float* __restrict__ out) {
    int col = N0 + (lane & 15);
    int hi8 = (lane & 16) ? 8 : 0;
    float bv = bias[col];
#pragma unroll
    for (int j = 0; j < 8; ++j) {
        int row = M0 + j + hi8;
        out[(size_t)row * D_MODEL + col] = c[j] + bv;
    }
}

__global__ __launch_bounds__(256) void gemm_out_kernel(
        const unsigned short* __restrict__ ctx,
        const unsigned short* __restrict__ wb,
        const float* __restrict__ bias,
        float* __restrict__ out) {
    const int lane  = threadIdx.x & 31;
    const int gwave = (blockIdx.x * blockDim.x + threadIdx.x) >> 5;
    const int NT = D_MODEL / 64;        // 16
    const int mt = gwave / NT;
    const int nt = gwave % NT;
    const int M0 = mt * 64, N0 = nt * 64;

    v8f acc[16];
#pragma unroll
    for (int i = 0; i < 16; ++i) acc[i] = {};

#pragma unroll 2
    for (int k0 = 0; k0 < D_MODEL; k0 += 32) {
        v16bf af[4], bf[4];
#pragma unroll
        for (int i = 0; i < 4; ++i)
            af[i] = load_a(ctx, D_MODEL, M0 + 16 * i, k0, lane);
#pragma unroll
        for (int i = 0; i < 4; ++i)
            bf[i] = load_b(wb, D_MODEL, N0 + 16 * i, k0, lane);
#pragma unroll
        for (int i = 0; i < 4; ++i)
#pragma unroll
            for (int j = 0; j < 4; ++j)
                acc[i * 4 + j] = wmma_bf(af[i], bf[j], acc[i * 4 + j]);
    }
#pragma unroll
    for (int i = 0; i < 4; ++i)
#pragma unroll
        for (int j = 0; j < 4; ++j)
            store_out_tile(acc[i * 4 + j], M0 + 16 * i, N0 + 16 * j,
                           lane, bias, out);
}

// ---------------------------------------------------------------------------
// Host-side launcher
// ---------------------------------------------------------------------------
extern "C" void kernel_launch(void* const* d_in, const int* in_sizes, int n_in,
                              void* d_out, int out_size, void* d_ws, size_t ws_size,
                              hipStream_t stream) {
    const float* x     = (const float*)d_in[0];
    const float* qkv_w = (const float*)d_in[1];
    const float* qkv_b = (const float*)d_in[2];
    const float* out_w = (const float*)d_in[3];
    const float* out_b = (const float*)d_in[4];
    float* out = (float*)d_out;

    char* ws = (char*)d_ws;
    // workspace layout (bytes), all 256-aligned
    unsigned short* xb  = (unsigned short*)(ws + 0);          //  8 MiB  x bf16
    unsigned short* qwb = (unsigned short*)(ws + 8388608);    //  6 MiB  qkv_w bf16
    unsigned short* owb = (unsigned short*)(ws + 14680064);   //  2 MiB  out_w bf16
    unsigned short* qb  = (unsigned short*)(ws + 16777216);   //  8 MiB  Q  [bh][s][d]
    unsigned short* kb  = (unsigned short*)(ws + 25165824);   //  8 MiB  K  [bh][s][d]
    unsigned short* vtb = (unsigned short*)(ws + 33554432);   //  8 MiB  V^T[bh][d][s]
    unsigned short* ctx = (unsigned short*)(ws + 41943040);   //  8 MiB  attn output bf16

    // 1) convert to bf16: (4096+3072+1024)*1024 elements
    hipLaunchKernelGGL(convert_kernel, dim3(32768), dim3(256), 0, stream,
                       x, qkv_w, out_w, xb, qwb, owb);
    // 2) QKV projection: 64 x 48 tiles of 64x64, 8 waves/block
    hipLaunchKernelGGL(gemm_qkv_kernel, dim3(384), dim3(256), 0, stream,
                       xb, qwb, qkv_b, qb, kb, vtb);
    // 3) flash attention: 32 (b,h) x 16 row-groups = 512 blocks
    hipLaunchKernelGGL(attn_kernel, dim3(512), dim3(256), 0, stream,
                       qb, kb, vtb, ctx);
    // 4) output projection: 64 x 16 tiles of 64x64
    hipLaunchKernelGGL(gemm_out_kernel, dim3(128), dim3(256), 0, stream,
                       ctx, owb, out_b, out);
}
